// PredictiveCodingEEG_64639257805093
// MI455X (gfx1250) — matrix operand order repaired and checked by reading, not compiled, not run on hardware
//
#include <hip/hip_runtime.h>
#include <hip/hip_bf16.h>

constexpr int FLAT = 16000;
constexpr int HID  = 128;
constexpr int EMB  = 64;
constexpr int NWIN = 4096;
constexpr int G3   = 384;   // 3*HID

typedef __attribute__((ext_vector_type(16))) _Float16 v16h;
typedef __attribute__((ext_vector_type(8)))  float    v8f;

// D = A*B + C, f16 inputs, f32 accumulate (V_WMMA_F32_16X16X32_F16)
__device__ __forceinline__ v8f wmma16(v16h a, v16h b, v8f c) {
  return __builtin_amdgcn_wmma_f32_16x16x32_f16(
      /*neg_a=*/false, a, /*neg_b=*/false, b,
      /*c_mod=*/(short)0, c, /*reuse_a=*/false, /*reuse_b=*/false);
}

// Build a 16-half fragment from two aligned runs of 8 f32 (cvt f32->f16).
__device__ __forceinline__ v16h frag_g(const float* __restrict__ p0,
                                       const float* __restrict__ p1) {
  const float4* q0 = reinterpret_cast<const float4*>(p0);
  const float4* q1 = reinterpret_cast<const float4*>(p1);
  float4 x0 = q0[0], x1 = q0[1], y0 = q1[0], y1 = q1[1];
  v16h f;
  f[0]=(_Float16)x0.x; f[1]=(_Float16)x0.y; f[2]=(_Float16)x0.z; f[3]=(_Float16)x0.w;
  f[4]=(_Float16)x1.x; f[5]=(_Float16)x1.y; f[6]=(_Float16)x1.z; f[7]=(_Float16)x1.w;
  f[8]=(_Float16)y0.x; f[9]=(_Float16)y0.y; f[10]=(_Float16)y0.z; f[11]=(_Float16)y0.w;
  f[12]=(_Float16)y1.x; f[13]=(_Float16)y1.y; f[14]=(_Float16)y1.z; f[15]=(_Float16)y1.w;
  return f;
}

__device__ __forceinline__ v16h frag_lds(const _Float16* p0, const _Float16* p1) {
  v16h f;
  #pragma unroll
  for (int j = 0; j < 8; ++j) f[j] = p0[j];
  #pragma unroll
  for (int j = 0; j < 8; ++j) f[8 + j] = p1[j];
  return f;
}

// ---------------------------------------------------------------------------
// Kernel 1: E = relu(flat @ W_enc.T + b_enc)  fused with  gi = E @ W_ih.T + b_ih
// grid.x = NWIN/16 (M tiles), 256 threads = 8 waves covering N=128 (HID).
// K-loop unrolled x2 so each wave keeps two independent b128 load batches in
// flight (software pipelining across the cvt+WMMA tail of the prior step).
// ---------------------------------------------------------------------------
__global__ __launch_bounds__(256) void enc_gi_kernel(
    const float* __restrict__ flat, const float* __restrict__ W_enc,
    const float* __restrict__ b_enc, const float* __restrict__ W_ih,
    const float* __restrict__ b_ih, float* __restrict__ gi)
{
  __shared__ _Float16 e_sh[16][HID + 8];   // E tile (f16), padded vs bank conflicts
  const int tid  = threadIdx.x;
  const int wave = tid >> 5;
  const int lane = tid & 31;
  const int col  = lane & 15;
  const int hi   = lane >> 4;
  const int m0   = blockIdx.x * 16;
  const int akoff = hi * 8;    // A fragment K base per half-wave
  const int bkoff = hi * 8;    // B fragment K base per half-wave

  // Phase 1: encoder GEMM, K = 16000 in steps of 32, unrolled x2 (500 = 2*250)
  const int n0 = wave * 16;
  const float* arow = flat  + (size_t)(m0 + col) * FLAT;
  const float* brow = W_enc + (size_t)(n0 + col) * FLAT;
  v8f c = {};
  #pragma unroll 1
  for (int k0 = 0; k0 < FLAT; k0 += 64) {
    v16h a0 = frag_g(arow + k0 + akoff,      arow + k0 + akoff + 16);
    v16h b0 = frag_g(brow + k0 + bkoff,      brow + k0 + bkoff + 16);
    v16h a1 = frag_g(arow + k0 + 32 + akoff, arow + k0 + 32 + akoff + 16);
    v16h b1 = frag_g(brow + k0 + 32 + bkoff, brow + k0 + 32 + bkoff + 16);
    c = wmma16(a0, b0, c);
    c = wmma16(a1, b1, c);
  }
  const float be = b_enc[n0 + col];
  #pragma unroll
  for (int r = 0; r < 8; ++r) {
    float e = c[r] + be;
    e_sh[r + 8 * hi][n0 + col] = (_Float16)(e > 0.f ? e : 0.f);
  }
  __syncthreads();

  // Phase 2: gi = E @ W_ih.T + b_ih  (N = 384 -> 24 tiles, 3 per wave)
  #pragma unroll
  for (int t = 0; t < 3; ++t) {
    const int j0 = (wave + t * 8) * 16;
    const float* wr = W_ih + (size_t)(j0 + col) * HID;
    v8f cg = {};
    #pragma unroll
    for (int kk = 0; kk < 4; ++kk) {
      const _Float16* er = &e_sh[col][kk * 32 + akoff];
      v16h a = frag_lds(er, er + 16);
      v16h b = frag_g(wr + kk * 32 + bkoff, wr + kk * 32 + bkoff + 16);
      cg = wmma16(a, b, cg);
    }
    const float bi = b_ih[j0 + col];
    #pragma unroll
    for (int r = 0; r < 8; ++r)
      gi[(size_t)(m0 + r + 8 * hi) * G3 + j0 + col] = cg[r] + bi;
  }
}

// ---------------------------------------------------------------------------
// Kernel 2: sequential GRU scan. One workgroup, 384 threads; each lane keeps
// its W_hh row fully register-resident; h lives in LDS (broadcast reads).
// ---------------------------------------------------------------------------
__global__ __launch_bounds__(384) void gru_kernel(
    const float* __restrict__ gi, const float* __restrict__ W_hh,
    const float* __restrict__ b_hh, float* __restrict__ H)
{
  __shared__ float h_sh[HID];
  __shared__ float ghs[G3];
  __shared__ float gsh[G3];
  const int j = threadIdx.x;            // 0..383
  float w[HID];
  #pragma unroll
  for (int k = 0; k < HID; ++k) w[k] = W_hh[(size_t)j * HID + k];
  const float bh = b_hh[j];
  if (j < HID) h_sh[j] = 0.f;
  __syncthreads();

  for (int t = 0; t < NWIN; ++t) {
    const float g = gi[(size_t)t * G3 + j];   // prefetch; latency hidden by dot
    float acc = bh;
    #pragma unroll
    for (int k = 0; k < HID; ++k) acc += w[k] * h_sh[k];
    ghs[j] = acc;
    gsh[j] = g;
    __syncthreads();
    if (j < HID) {
      const float r  = 1.f / (1.f + __expf(-(gsh[j] + ghs[j])));
      const float z  = 1.f / (1.f + __expf(-(gsh[HID + j] + ghs[HID + j])));
      const float n  = tanhf(gsh[2 * HID + j] + r * ghs[2 * HID + j]);
      const float hn = (1.f - z) * n + z * h_sh[j];
      h_sh[j] = hn;
      H[(size_t)t * HID + j] = hn;
    }
    __syncthreads();
  }
}

// ---------------------------------------------------------------------------
// Kernel 3: pc_emb = H @ W_pc.T + b_pc  (4096 x 64, K=128)
// ---------------------------------------------------------------------------
__global__ __launch_bounds__(128) void pc_kernel(
    const float* __restrict__ H, const float* __restrict__ W_pc,
    const float* __restrict__ b_pc, float* __restrict__ out)
{
  const int wave = threadIdx.x >> 5, lane = threadIdx.x & 31;
  const int col = lane & 15, hi = lane >> 4;
  const int m0 = blockIdx.x * 16;
  const int n0 = wave * 16;
  const int akoff = hi * 8, bkoff = hi * 8;
  const float* ar = H    + (size_t)(m0 + col) * HID;
  const float* br = W_pc + (size_t)(n0 + col) * HID;
  v8f c = {};
  #pragma unroll
  for (int kk = 0; kk < 4; ++kk) {
    v16h a = frag_g(ar + kk * 32 + akoff, ar + kk * 32 + akoff + 16);
    v16h b = frag_g(br + kk * 32 + bkoff, br + kk * 32 + bkoff + 16);
    c = wmma16(a, b, c);
  }
  const float bias = b_pc[n0 + col];
  #pragma unroll
  for (int r = 0; r < 8; ++r)
    out[(size_t)(m0 + r + 8 * hi) * EMB + n0 + col] = c[r] + bias;
}

// ---------------------------------------------------------------------------
// Kernel 4: fused pred + MSE. Never materializes pred (saves 262 MB of HBM
// stores). A fragments (H rows) register-resident across all 1000 N-tiles.
// errs[w] accumulated via global_atomic_add_f32, pre-scaled by 1/FLAT.
// ---------------------------------------------------------------------------
__global__ __launch_bounds__(256) void pred_err_kernel(
    const float* __restrict__ H, const float* __restrict__ W_pred,
    const float* __restrict__ b_pred, const float* __restrict__ flat,
    float* __restrict__ err)
{
  const int wave = threadIdx.x >> 5, lane = threadIdx.x & 31;
  const int col = lane & 15, hi = lane >> 4;
  const int m0 = blockIdx.x * 16;
  const int akoff = hi * 8, bkoff = hi * 8;

  const float* ar = H + (size_t)(m0 + col) * HID;
  v16h A[4];
  #pragma unroll
  for (int kk = 0; kk < 4; ++kk)
    A[kk] = frag_g(ar + kk * 32 + akoff, ar + kk * 32 + akoff + 16);

  float acc[8];
  #pragma unroll
  for (int r = 0; r < 8; ++r) acc[r] = 0.f;

  for (int nt = wave; nt < FLAT / 16; nt += 8) {
    const int f0 = nt * 16;
    const float* br = W_pred + (size_t)(f0 + col) * HID;
    v8f c = {};
    #pragma unroll
    for (int kk = 0; kk < 4; ++kk) {
      v16h b = frag_g(br + kk * 32 + bkoff, br + kk * 32 + bkoff + 16);
      c = wmma16(A[kk], b, c);
    }
    const float bp = b_pred[f0 + col];
    #pragma unroll
    for (int r = 0; r < 8; ++r) {
      const int w  = m0 + r + 8 * hi;
      const int wl = (w <= NWIN - 2) ? w : (NWIN - 2);   // clamp for safe load
      const float tgt = flat[(size_t)(wl + 1) * FLAT + f0 + col];
      float d = c[r] + bp - tgt;
      d = (w <= NWIN - 2) ? d : 0.f;                     // mask row w=4095
      acc[r] += d * d;
    }
  }
  // reduce across the 16 lanes of each half-wave (each half owns 8 rows)
  #pragma unroll
  for (int r = 0; r < 8; ++r) {
    float v = acc[r];
    #pragma unroll
    for (int s = 1; s < 16; s <<= 1) v += __shfl_xor(v, s, 32);
    if (col == 0) {
      const int w = m0 + r + 8 * hi;
      if (w <= NWIN - 2) atomicAdd(&err[w], v * (1.0f / FLAT));
    }
  }
}

__global__ void zero_err_kernel(float* __restrict__ err) {
  const int i = blockIdx.x * blockDim.x + threadIdx.x;
  if (i < NWIN) err[i] = 0.f;   // also yields pred_err[4095] == 0
}

extern "C" void kernel_launch(void* const* d_in, const int* in_sizes, int n_in,
                              void* d_out, int out_size, void* d_ws, size_t ws_size,
                              hipStream_t stream) {
  (void)in_sizes; (void)n_in; (void)out_size; (void)ws_size;
  const float* windows = (const float*)d_in[0];   // (4096, 64, 250) == (4096, 16000)
  const float* W_enc   = (const float*)d_in[1];
  const float* b_enc   = (const float*)d_in[2];
  const float* W_ih    = (const float*)d_in[3];
  const float* W_hh    = (const float*)d_in[4];
  const float* b_ih    = (const float*)d_in[5];
  const float* b_hh    = (const float*)d_in[6];
  const float* W_pred  = (const float*)d_in[7];
  const float* b_pred  = (const float*)d_in[8];
  const float* W_pc    = (const float*)d_in[9];
  const float* b_pc    = (const float*)d_in[10];

  float* out  = (float*)d_out;
  float* err  = out + (size_t)NWIN * EMB;         // pred_err region
  float* gi   = (float*)d_ws;                     // 4096*384 f32
  float* Hbuf = gi + (size_t)NWIN * G3;           // 4096*128 f32  (~8.4 MB total)

  zero_err_kernel<<<(NWIN + 255) / 256, 256, 0, stream>>>(err);
  enc_gi_kernel<<<NWIN / 16, 256, 0, stream>>>(windows, W_enc, b_enc, W_ih, b_ih, gi);
  gru_kernel<<<1, 384, 0, stream>>>(gi, W_hh, b_hh, Hbuf);
  pc_kernel<<<NWIN / 16, 128, 0, stream>>>(Hbuf, W_pc, b_pc, out);
  pred_err_kernel<<<NWIN / 16, 256, 0, stream>>>(Hbuf, W_pred, b_pred, windows, err);
}
